// MultiHeadAttention_88467736363117
// MI455X (gfx1250) — compile-verified, matrix operand order checked by reading, not run on hardware
//
#include <hip/hip_runtime.h>
#include <hip/hip_bf16.h>

// ---------------------------------------------------------------------------
// MHA for MI455X (gfx1250).
//  - bf16 WMMA (V_WMMA_F32_16X16X32_BF16) for all GEMMs, fp32 accumulate.
//  - LDS-free GEMM fragment loads: every operand contiguous 128-bit, L2-resident.
//  - Fused flash-style scores+softmax+AV: online softmax in registers,
//    scores recomputed in pass 2 (compute is free, HBM traffic is not),
//    probabilities written to d_out exactly once.
// ---------------------------------------------------------------------------

typedef __attribute__((ext_vector_type(8)))  __bf16       v8bf;
typedef __attribute__((ext_vector_type(16))) __bf16       v16bf;
typedef __attribute__((ext_vector_type(8)))  float        v8f;
typedef __attribute__((ext_vector_type(4)))  float        f4;
typedef __attribute__((ext_vector_type(4)))  unsigned int u4;

#define B_   4
#define H_   8
#define T_   2048
#define S_   2048
#define E_   512
#define DH_  64
#define NEG_INF_ (-1e30f)

static __device__ __forceinline__ __bf16 tobf(float f) { return (__bf16)f; }

static __device__ __forceinline__ v8bf load8bf(const __bf16* p) {
    return __builtin_bit_cast(v8bf, *(const u4*)p);
}
static __device__ __forceinline__ v8bf cvt8(const float* p) {
    f4 x = *(const f4*)p;
    f4 y = *(const f4*)(p + 4);
    v8bf r;
    r[0] = (__bf16)x[0]; r[1] = (__bf16)x[1]; r[2] = (__bf16)x[2]; r[3] = (__bf16)x[3];
    r[4] = (__bf16)y[0]; r[5] = (__bf16)y[1]; r[6] = (__bf16)y[2]; r[7] = (__bf16)y[3];
    return r;
}
static __device__ __forceinline__ v16bf cat(v8bf lo, v8bf hi) {
    return __builtin_shufflevector(lo, hi, 0,1,2,3,4,5,6,7,8,9,10,11,12,13,14,15);
}
static __device__ __forceinline__ v8f wmma_bf16(v16bf a, v16bf b, v8f c) {
    return __builtin_amdgcn_wmma_f32_16x16x32_bf16(false, a, false, b,
                                                   (short)0, c, false, false);
}

// ---------------------------------------------------------------------------
// Weight transpose: Wt[n][k] = bf16(W[k][n]).  32x32 LDS tiles.
// ---------------------------------------------------------------------------
__global__ __launch_bounds__(256)
void wtrans_kernel(const float* __restrict__ W, __bf16* __restrict__ Wt)
{
    __shared__ float tile[32][33];
    const int bx = blockIdx.x % (E_ / 32), by = blockIdx.x / (E_ / 32);
    const int k0 = by * 32, n0 = bx * 32;
    const int tx = threadIdx.x & 31, ty = threadIdx.x >> 5;
    #pragma unroll
    for (int i = 0; i < 32; i += 8)
        tile[ty + i][tx] = W[(size_t)(k0 + ty + i) * E_ + n0 + tx];
    __syncthreads();
    #pragma unroll
    for (int i = 0; i < 32; i += 8)
        Wt[(size_t)(n0 + ty + i) * E_ + k0 + tx] = tobf(tile[tx][ty + i]);
}

// ---------------------------------------------------------------------------
// Projection GEMM (LDS-free): row = [b,t], n = h*64+d
//   vmode 0: out[b,h,t,d] (Q, K)      vmode 1: out[b,h,d,t] (V, d-major)
// grid.x = (B*T/16)*(E/128), block 256 (8 waves, one 16x16 tile each)
// ---------------------------------------------------------------------------
__global__ __launch_bounds__(256)
void proj_kernel(const float* __restrict__ X, const __bf16* __restrict__ Wt,
                 const float* __restrict__ bias, __bf16* __restrict__ out,
                 float scale, int vmode)
{
    const int tid = threadIdx.x, lane = tid & 31, wv = tid >> 5;
    const int nStrips = E_ / 128;
    const int m0 = (blockIdx.x / nStrips) * 16;
    const int nb = (blockIdx.x % nStrips) * 128;
    const int rl = lane & 15;
    const int ko = (lane < 16) ? 0 : 8;                 // ISA 16-bit operand layout
    const int n  = nb + (wv << 4) + rl;

    const float*  arow = X  + (size_t)(m0 + rl) * E_;
    const __bf16* brow = Wt + (size_t)n * E_;

    v8f acc = {};
    #pragma unroll 4
    for (int kk = 0; kk < E_; kk += 32) {
        __builtin_prefetch(arow + kk + 128, 0, 0);
        v16bf a = cat(cvt8(arow + kk + ko),    cvt8(arow + kk + 16 + ko));
        v16bf b = cat(load8bf(brow + kk + ko), load8bf(brow + kk + 16 + ko));
        acc = wmma_bf16(a, b, acc);
    }

    const float bval = bias[n];
    const int h = n >> 6, d = n & 63;
    #pragma unroll
    for (int j = 0; j < 8; ++j) {
        int row = m0 + j + ((lane < 16) ? 0 : 8);       // C/D layout: VGPR j
        int bb  = row / T_, t = row % T_;
        __bf16 v = tobf((acc[j] + bval) * scale);
        if (vmode == 0)
            out[((size_t)(bb * H_ + h) * T_ + t) * DH_ + d] = v;
        else
            out[((size_t)(bb * H_ + h) * DH_ + d) * S_ + t] = v;
    }
}

// ---------------------------------------------------------------------------
// Fused attention: per wave = one 16-row T-tile of one (b,h).
//  Pass 1: sweep S with QK^T WMMAs, online per-lane softmax stats.
//  Merge stats across the 16 lanes sharing each row (shfl_xor tree).
//  Pass 2: re-sweep S, emit normalized P (fp32 -> d_out, once), bounce P tile
//          through wave-private LDS into A-layout, 4 AV WMMAs per 32-wide step.
// grid.x = T/16/8, grid.y = B*H, block 256 (8 independent waves)
// ---------------------------------------------------------------------------
__global__ __launch_bounds__(256)
void fused_attn_kernel(const __bf16* __restrict__ Q, const __bf16* __restrict__ K,
                       const __bf16* __restrict__ Vt,
                       const unsigned char* __restrict__ attn_mask,
                       const unsigned char* __restrict__ kpm,
                       float* __restrict__ Pout, __bf16* __restrict__ Abf)
{
    __shared__ __bf16 pbuf[8][16][32];                  // 1KB per wave, wave-private

    const int tid = threadIdx.x, lane = tid & 31, wv = tid >> 5;
    const int bh = blockIdx.y, b = bh >> 3, h = bh & 7;
    const int t0 = (blockIdx.x * 8 + wv) * 16;
    const int rl = lane & 15;
    const int ko = (lane < 16) ? 0 : 8;
    const int jbase = (lane < 16) ? 0 : 8;

    // Q fragments for this tile (Dh = 64 -> two resident A-frags)
    const __bf16* qrow = Q + ((size_t)bh * T_ + t0 + rl) * DH_;
    const v16bf qa0 = cat(load8bf(qrow + ko),      load8bf(qrow + 16 + ko));
    const v16bf qa1 = cat(load8bf(qrow + 32 + ko), load8bf(qrow + 48 + ko));

    // ---------------- Pass 1: online softmax stats --------------------------
    float m[8], ssum[8];
    #pragma unroll
    for (int j = 0; j < 8; ++j) { m[j] = -3.4e38f; ssum[j] = 0.f; }

    for (int st = 0; st < S_; st += 16) {
        const __bf16* krow = K + ((size_t)bh * S_ + st + rl) * DH_;
        v16bf kb0 = cat(load8bf(krow + ko),      load8bf(krow + 16 + ko));
        v16bf kb1 = cat(load8bf(krow + 32 + ko), load8bf(krow + 48 + ko));
        v8f acc = {};
        acc = wmma_bf16(qa0, kb0, acc);
        acc = wmma_bf16(qa1, kb1, acc);

        const int s = st + rl;
        const unsigned char kp = kpm[b * S_ + s];
        #pragma unroll
        for (int j = 0; j < 8; ++j) {
            int t = t0 + j + jbase;
            float v = acc[j];
            if (attn_mask[(size_t)t * S_ + s] | kp) v = NEG_INF_;
            float nm = fmaxf(m[j], v);
            ssum[j] = ssum[j] * __expf(m[j] - nm) + __expf(v - nm);
            m[j] = nm;
        }
    }

    // merge the 16 lanes that share each row (xor masks 1,2,4,8 stay in half)
    #pragma unroll
    for (int mask = 1; mask < 16; mask <<= 1) {
        #pragma unroll
        for (int j = 0; j < 8; ++j) {
            float om = __shfl_xor(m[j],    mask, 32);
            float os = __shfl_xor(ssum[j], mask, 32);
            float nm = fmaxf(m[j], om);
            ssum[j] = ssum[j] * __expf(m[j] - nm) + os * __expf(om - nm);
            m[j] = nm;
        }
    }
    float inv[8];
    #pragma unroll
    for (int j = 0; j < 8; ++j) inv[j] = 1.0f / ssum[j];

    // ---------------- Pass 2: emit P once + AV accumulate -------------------
    v8f accd0 = {}, accd1 = {}, accd2 = {}, accd3 = {};

    for (int st = 0; st < S_; st += 32) {
        // two 16-wide sub-tiles -> one 16x32 probability tile in pbuf
        #pragma unroll
        for (int half = 0; half < 2; ++half) {
            const int stt = st + half * 16;
            const __bf16* krow = K + ((size_t)bh * S_ + stt + rl) * DH_;
            v16bf kb0 = cat(load8bf(krow + ko),      load8bf(krow + 16 + ko));
            v16bf kb1 = cat(load8bf(krow + 32 + ko), load8bf(krow + 48 + ko));
            v8f acc = {};
            acc = wmma_bf16(qa0, kb0, acc);
            acc = wmma_bf16(qa1, kb1, acc);

            const int s = stt + rl;
            const unsigned char kp = kpm[b * S_ + s];
            #pragma unroll
            for (int j = 0; j < 8; ++j) {
                int t = t0 + j + jbase;
                float v = acc[j];
                if (attn_mask[(size_t)t * S_ + s] | kp) v = NEG_INF_;
                float p = __expf(v - m[j]) * inv[j];
                Pout[((size_t)bh * T_ + t) * S_ + s] = p;          // the one write
                pbuf[wv][j + jbase][half * 16 + rl] = tobf(p);
            }
        }

        // P tile: C-layout -> A-layout via wave-private LDS bounce
        v16bf pa = cat(load8bf(&pbuf[wv][rl][ko]),
                       load8bf(&pbuf[wv][rl][16 + ko]));

        const __bf16* vbase = Vt + ((size_t)bh * DH_ + rl) * S_ + st;
        v16bf vb;
        vb = cat(load8bf(vbase + ko), load8bf(vbase + 16 + ko));
        accd0 = wmma_bf16(pa, vb, accd0);
        vb = cat(load8bf(vbase + 16 * S_ + ko), load8bf(vbase + 16 * S_ + 16 + ko));
        accd1 = wmma_bf16(pa, vb, accd1);
        vb = cat(load8bf(vbase + 32 * S_ + ko), load8bf(vbase + 32 * S_ + 16 + ko));
        accd2 = wmma_bf16(pa, vb, accd2);
        vb = cat(load8bf(vbase + 48 * S_ + ko), load8bf(vbase + 48 * S_ + 16 + ko));
        accd3 = wmma_bf16(pa, vb, accd3);
    }

    // store attention output tile (16 x 64) as bf16 [B,T,E]
    #pragma unroll
    for (int j = 0; j < 8; ++j) {
        int t = t0 + j + jbase;
        __bf16* orow = Abf + ((size_t)(b * T_ + t)) * E_ + h * DH_;
        orow[rl]      = tobf(accd0[j]);
        orow[16 + rl] = tobf(accd1[j]);
        orow[32 + rl] = tobf(accd2[j]);
        orow[48 + rl] = tobf(accd3[j]);
    }
}

// ---------------------------------------------------------------------------
// Output projection: out[row,n] = A_bf[row,:] @ WoT[n,:] + bo[n].  LDS-free.
// ---------------------------------------------------------------------------
__global__ __launch_bounds__(256)
void outproj_kernel(const __bf16* __restrict__ A, const __bf16* __restrict__ Wt,
                    const float* __restrict__ bias, float* __restrict__ out)
{
    const int tid = threadIdx.x, lane = tid & 31, wv = tid >> 5;
    const int nStrips = E_ / 128;
    const int m0 = (blockIdx.x / nStrips) * 16;
    const int nb = (blockIdx.x % nStrips) * 128;
    const int rl = lane & 15;
    const int ko = (lane < 16) ? 0 : 8;
    const int n  = nb + (wv << 4) + rl;

    const __bf16* arow = A  + (size_t)(m0 + rl) * E_;
    const __bf16* brow = Wt + (size_t)n * E_;

    v8f acc = {};
    #pragma unroll 4
    for (int kk = 0; kk < E_; kk += 32) {
        v16bf a = cat(load8bf(arow + kk + ko), load8bf(arow + kk + 16 + ko));
        v16bf b = cat(load8bf(brow + kk + ko), load8bf(brow + kk + 16 + ko));
        acc = wmma_bf16(a, b, acc);
    }
    const float bval = bias[n];
    #pragma unroll
    for (int j = 0; j < 8; ++j) {
        int row = m0 + j + ((lane < 16) ? 0 : 8);
        out[(size_t)row * E_ + n] = acc[j] + bval;
    }
}

// ---------------------------------------------------------------------------
extern "C" void kernel_launch(void* const* d_in, const int* in_sizes, int n_in,
                              void* d_out, int out_size, void* d_ws, size_t ws_size,
                              hipStream_t stream)
{
    (void)in_sizes; (void)n_in; (void)out_size; (void)ws_size;

    const float* query = (const float*)d_in[0];
    const float* key   = (const float*)d_in[1];
    const float* value = (const float*)d_in[2];
    const unsigned char* kpm   = (const unsigned char*)d_in[3];
    const unsigned char* amask = (const unsigned char*)d_in[4];
    const float* Wq = (const float*)d_in[5];
    const float* bq = (const float*)d_in[6];
    const float* Wk = (const float*)d_in[7];
    const float* bk = (const float*)d_in[8];
    const float* Wv = (const float*)d_in[9];
    const float* bv = (const float*)d_in[10];
    const float* Wo = (const float*)d_in[11];
    const float* bo = (const float*)d_in[12];

    float* out   = (float*)d_out;                       // [B,T,E]
    float* attnW = out + (size_t)B_ * T_ * E_;          // [B,H,T,S]

    const size_t QKV = (size_t)B_ * H_ * T_ * DH_;
    const size_t WT  = (size_t)E_ * E_;
    __bf16* Qbf = (__bf16*)d_ws;
    __bf16* Kbf = Qbf + QKV;
    __bf16* Vtb = Kbf + QKV;                            // [b,h,d,s] (d-major)
    __bf16* Abf = Vtb + QKV;                            // [B,T,E]
    __bf16* WqT = Abf + QKV;
    __bf16* WkT = WqT + WT;
    __bf16* WvT = WkT + WT;
    __bf16* WoT = WvT + WT;

    const float scale = 0.125f;                         // 64^-0.5

    dim3 gt((E_ / 32) * (E_ / 32));
    wtrans_kernel<<<gt, 256, 0, stream>>>(Wq, WqT);
    wtrans_kernel<<<gt, 256, 0, stream>>>(Wk, WkT);
    wtrans_kernel<<<gt, 256, 0, stream>>>(Wv, WvT);
    wtrans_kernel<<<gt, 256, 0, stream>>>(Wo, WoT);

    dim3 gp((B_ * T_ / 16) * (E_ / 128));
    proj_kernel<<<gp, 256, 0, stream>>>(query, WqT, bq, Qbf, scale, 0);
    proj_kernel<<<gp, 256, 0, stream>>>(key,   WkT, bk, Kbf, 1.0f, 0);
    proj_kernel<<<gp, 256, 0, stream>>>(value, WvT, bv, Vtb, 1.0f, 1);

    dim3 ga(T_ / 16 / 8, B_ * H_);                      // 16 x 32 blocks, 8 waves
    fused_attn_kernel<<<ga, 256, 0, stream>>>(Qbf, Kbf, Vtb, amask, kpm,
                                              attnW, Abf);

    outproj_kernel<<<gp, 256, 0, stream>>>(Abf, WoT, bo, out);
}
